// GAT_43061342110391
// MI455X (gfx1250) — compile-verified, matrix operand order checked by reading, not run on hardware
//
#include <hip/hip_runtime.h>
#include <hip/hip_bf16.h>

// ---------------------------------------------------------------------------
// GAT (2-layer, PyG-style) for MI455X / gfx1250.
//  - Dense GEMMs via v_wmma_f32_16x16x32_f16 (f16 operands, f32 accum).
//    B (weights) pre-packed into per-lane fragment order -> b128 loads only.
//    Each wave keeps 3-4 accumulators and reuses one A fragment per k-step.
//  - Edge softmax/aggregation via L2 atomics (h1 = 51MB << 192MB L2).
// ---------------------------------------------------------------------------

#define N_NODES   50000
#define E_EDGES   800000
#define F_IN      128
#define HID       64
#define HEADS     4
#define NCLS      47
#define NCLS_PAD  48
#define NEG_SLOPE 0.2f

typedef __attribute__((ext_vector_type(16))) _Float16 v16h;
typedef __attribute__((ext_vector_type(8)))  float    v8f;

// ------------------------------ utilities ----------------------------------

__global__ void fill_f32(float* __restrict__ p, float v, int n) {
    int i = blockIdx.x * blockDim.x + threadIdx.x;
    if (i < n) p[i] = v;
}

// Pack W[K, cols] (f32 row-major) into WMMA B-fragment order (f16):
//   Bp[((kt*ntiles + nt)*32 + lane)*16 + j]
// following the CDNA5 16-bit B 32x16 layout: lane L holds col N = L&15,
// K-offset 8*(L>>4); j=0..7 -> K+0..7, j=8..15 -> K+16..23.
__global__ void pack_b_f32(const float* __restrict__ W, _Float16* __restrict__ Bp,
                           int K, int cols, int ntiles) {
    int t = blockIdx.x * blockDim.x + threadIdx.x;
    int total = (K >> 5) * ntiles * 32;
    if (t >= total) return;
    int lane = t & 31;
    int nt   = (t >> 5) % ntiles;
    int kt   = (t >> 5) / ntiles;
    int ncol = nt * 16 + (lane & 15);
    int koff = (lane >> 4) * 8;
    _Float16* dst = Bp + (size_t)t * 16;
#pragma unroll
    for (int j = 0; j < 8; ++j) {
        int k  = kt * 32 + koff + j;
        int k2 = k + 16;
        dst[j]     = (ncol < cols) ? (_Float16)W[(size_t)k  * cols + ncol] : (_Float16)0.f;
        dst[8 + j] = (ncol < cols) ? (_Float16)W[(size_t)k2 * cols + ncol] : (_Float16)0.f;
    }
}

// ------------------------------ WMMA GEMM ----------------------------------
// C[M, ldc] = A[M, K] (f32, cvt on the fly) x Bp (pre-packed f16 fragments).
// One wave computes NT 16x16 D tiles sharing one A fragment per k-step.
template <int NT>
__global__ void __launch_bounds__(128)
wmma_gemm_packed(const float* __restrict__ A, const _Float16* __restrict__ Bp,
                 float* __restrict__ C, int M, int K, int ntilesTot,
                 int ncols, int ldc) {
    int tileM  = blockIdx.x;
    int ntBase = threadIdx.y * NT;
    if (ntBase >= ntilesTot) return;           // uniform per wave

    int lane = threadIdx.x;
    int mrow = tileM * 16 + (lane & 15);
    if (mrow >= M) mrow = M - 1;               // clamp (tiles exact for N=50000)
    int koff = (lane >> 4) * 8;
    const float* arow = A + (size_t)mrow * K;
    const v16h* bfrag = (const v16h*)Bp;

    v8f acc[NT];
#pragma unroll
    for (int i = 0; i < NT; ++i) acc[i] = (v8f){};

    int nkt = K >> 5;
    for (int kt = 0; kt < nkt; ++kt) {
        int k0 = kt * 32;
        // A fragment: 4 x b128 loads, convert f32 -> f16
        float4 a0 = *(const float4*)(arow + k0 + koff);
        float4 a1 = *(const float4*)(arow + k0 + koff + 4);
        float4 a2 = *(const float4*)(arow + k0 + koff + 16);
        float4 a3 = *(const float4*)(arow + k0 + koff + 20);
        v16h af;
        af[0]  = (_Float16)a0.x; af[1]  = (_Float16)a0.y;
        af[2]  = (_Float16)a0.z; af[3]  = (_Float16)a0.w;
        af[4]  = (_Float16)a1.x; af[5]  = (_Float16)a1.y;
        af[6]  = (_Float16)a1.z; af[7]  = (_Float16)a1.w;
        af[8]  = (_Float16)a2.x; af[9]  = (_Float16)a2.y;
        af[10] = (_Float16)a2.z; af[11] = (_Float16)a2.w;
        af[12] = (_Float16)a3.x; af[13] = (_Float16)a3.y;
        af[14] = (_Float16)a3.z; af[15] = (_Float16)a3.w;

        size_t fbase = ((size_t)kt * ntilesTot + ntBase) * 32 + lane;
#pragma unroll
        for (int i = 0; i < NT; ++i) {
            v16h bf = bfrag[fbase + (size_t)i * 32];   // 32B coalesced per lane
            acc[i] = __builtin_amdgcn_wmma_f32_16x16x32_f16(
                false, af, false, bf, (short)0, acc[i], false, false);
        }
    }

    int rbase = tileM * 16 + ((lane >> 4) << 3);
#pragma unroll
    for (int i = 0; i < NT; ++i) {
        int ncol = (ntBase + i) * 16 + (lane & 15);
        if (ncol < ncols) {
#pragma unroll
            for (int r = 0; r < 8; ++r) {
                int row = rbase + r;
                if (row < M) C[(size_t)row * ldc + ncol] = acc[i][r];
            }
        }
    }
}

// --------------------------- attention logits ------------------------------
__global__ void node_logits(const float* __restrict__ h,
                            const float* __restrict__ a_s,
                            const float* __restrict__ a_d,
                            float* __restrict__ als, float* __restrict__ ald,
                            int n, int heads, int ch, int ldh) {
    int t = blockIdx.x * blockDim.x + threadIdx.x;
    if (t >= n * heads) return;
    int node = t / heads, hd = t % heads;
    const float* hp = h + (size_t)node * ldh + hd * ch;
    const float* sp = a_s + hd * ch;
    const float* dp = a_d + hd * ch;
    float s = 0.f, d = 0.f;
    for (int c = 0; c < ch; ++c) {
        float v = hp[c];
        s += v * sp[c];
        d += v * dp[c];
    }
    als[t] = s;
    ald[t] = d;
}

// ------------------------------ edge kernels -------------------------------

__device__ inline void edge_sd(const int* __restrict__ ei, int e, int& s, int& d) {
    if (e < E_EDGES) { s = ei[e]; d = ei[E_EDGES + e]; }
    else             { s = e - E_EDGES; d = s; }        // self-loop
}

__device__ inline float atomic_max_f32(float* addr, float val) {
    if (val >= 0.f)
        return __int_as_float(atomicMax((int*)addr, __float_as_int(val)));
    return __uint_as_float(atomicMin((unsigned int*)addr, __float_as_uint(val)));
}

__global__ void edge_max_k(const int* __restrict__ ei,
                           const float* __restrict__ als,
                           const float* __restrict__ ald,
                           float* __restrict__ nmax, int heads, int etot) {
    int t = blockIdx.x * blockDim.x + threadIdx.x;
    if (t >= etot * heads) return;
    int e = t / heads, hd = t % heads;
    int s, d; edge_sd(ei, e, s, d);
    float v = als[s * heads + hd] + ald[d * heads + hd];
    v = (v >= 0.f) ? v : NEG_SLOPE * v;
    atomic_max_f32(&nmax[d * heads + hd], v);
}

__global__ void edge_exp_k(const int* __restrict__ ei,
                           const float* __restrict__ als,
                           const float* __restrict__ ald,
                           const float* __restrict__ nmax,
                           float* __restrict__ exb, float* __restrict__ den,
                           int heads, int etot) {
    int t = blockIdx.x * blockDim.x + threadIdx.x;
    if (t >= etot * heads) return;
    int e = t / heads, hd = t % heads;
    int s, d; edge_sd(ei, e, s, d);
    float v = als[s * heads + hd] + ald[d * heads + hd];
    v = (v >= 0.f) ? v : NEG_SLOPE * v;
    float ex = __expf(v - nmax[d * heads + hd]);
    exb[t] = ex;
    atomicAdd(&den[d * heads + hd], ex);
}

__global__ void edge_agg_k(const int* __restrict__ ei,
                           const float* __restrict__ exb,
                           const float* __restrict__ den,
                           const float* __restrict__ h,
                           float* __restrict__ agg,
                           int heads, int ch, int ldh, int lda, int etot) {
    int t = blockIdx.x * blockDim.x + threadIdx.x;
    if (t >= etot * heads) return;
    int e = t / heads, hd = t % heads;
    int s, d; edge_sd(ei, e, s, d);
    float alpha = exb[t] / (den[d * heads + hd] + 1e-16f);
    const float* hp = h + (size_t)s * ldh + hd * ch;
    float* op = agg + (size_t)d * lda + hd * ch;
    int c = 0;
    for (; c + 3 < ch; c += 4) {                 // b128 gather of h[src]
        float4 hv = *(const float4*)(hp + c);
        atomicAdd(&op[c + 0], alpha * hv.x);
        atomicAdd(&op[c + 1], alpha * hv.y);
        atomicAdd(&op[c + 2], alpha * hv.z);
        atomicAdd(&op[c + 3], alpha * hv.w);
    }
    for (; c < ch; ++c)
        atomicAdd(&op[c], alpha * hp[c]);
}

// --------------------------- epilogue kernels ------------------------------

__global__ void bias_elu_k(float* __restrict__ p, const float* __restrict__ b,
                           int n, int c) {
    int i = blockIdx.x * blockDim.x + threadIdx.x;
    if (i >= n * c) return;
    float v = p[i] + b[i % c];
    p[i] = (v > 0.f) ? v : (__expf(v) - 1.f);
}

// one wave32 per node: out = log_softmax(agg + b)
__global__ void log_softmax_k(const float* __restrict__ agg,
                              const float* __restrict__ b,
                              float* __restrict__ out, int n, int c) {
    int node = blockIdx.x * blockDim.y + threadIdx.y;
    if (node >= n) return;
    int lane = threadIdx.x;
    const float* ip = agg + (size_t)node * c;
    float m = -__builtin_inff();
    for (int j = lane; j < c; j += 32) m = fmaxf(m, ip[j] + b[j]);
#pragma unroll
    for (int off = 16; off > 0; off >>= 1) m = fmaxf(m, __shfl_xor(m, off));
    float s = 0.f;
    for (int j = lane; j < c; j += 32) s += __expf(ip[j] + b[j] - m);
#pragma unroll
    for (int off = 16; off > 0; off >>= 1) s += __shfl_xor(s, off);
    float ls = __logf(s);
    for (int j = lane; j < c; j += 32)
        out[(size_t)node * c + j] = ip[j] + b[j] - m - ls;
}

// ------------------------------ launcher -----------------------------------

extern "C" void kernel_launch(void* const* d_in, const int* in_sizes, int n_in,
                              void* d_out, int out_size, void* d_ws, size_t ws_size,
                              hipStream_t stream) {
    (void)in_sizes; (void)n_in; (void)out_size; (void)ws_size;

    const float* x   = (const float*)d_in[0];
    const int*   ei  = (const int*)d_in[1];
    const float* W1  = (const float*)d_in[2];
    const float* as1 = (const float*)d_in[3];
    const float* ad1 = (const float*)d_in[4];
    const float* b1  = (const float*)d_in[5];
    const float* W2  = (const float*)d_in[6];
    const float* as2 = (const float*)d_in[7];
    const float* ad2 = (const float*)d_in[8];
    const float* b2  = (const float*)d_in[9];
    float* out = (float*)d_out;

    const int ET  = E_EDGES + N_NODES;    // with self-loops
    const int HC1 = HEADS * HID;          // 256
    const int NT1 = HC1 / 16;             // 16 column tiles (layer 1)
    const int NT2 = NCLS_PAD / 16;        // 3 column tiles (layer 2)
    const int NKT1 = F_IN / 32;           // 4 k-steps
    const int NKT2 = HC1 / 32;            // 8 k-steps

    // workspace carving (256B aligned)
    char* w = (char*)d_ws;
    auto carve = [&](size_t bytes) {
        void* p = (void*)w;
        w += (bytes + 255) & ~(size_t)255;
        return p;
    };
    _Float16* Bp1  = (_Float16*)carve((size_t)NKT1 * NT1 * 32 * 16 * 2);  // 64KB
    _Float16* Bp2  = (_Float16*)carve((size_t)NKT2 * NT2 * 32 * 16 * 2);  // 24KB
    float* h1   = (float*)carve((size_t)N_NODES * HC1 * 4);
    float* als1 = (float*)carve((size_t)N_NODES * HEADS * 4);
    float* ald1 = (float*)carve((size_t)N_NODES * HEADS * 4);
    float* nm1  = (float*)carve((size_t)N_NODES * HEADS * 4);
    float* dn1  = (float*)carve((size_t)N_NODES * HEADS * 4);
    float* ex1  = (float*)carve((size_t)ET * HEADS * 4);
    float* agg1 = (float*)carve((size_t)N_NODES * HC1 * 4);   // reused as ELU out
    float* h2   = (float*)carve((size_t)N_NODES * NCLS_PAD * 4);
    float* als2 = (float*)carve((size_t)N_NODES * 4);
    float* ald2 = (float*)carve((size_t)N_NODES * 4);
    float* nm2  = (float*)carve((size_t)N_NODES * 4);
    float* dn2  = (float*)carve((size_t)N_NODES * 4);
    float* ex2  = (float*)carve((size_t)ET * 4);
    float* agg2 = (float*)carve((size_t)N_NODES * NCLS * 4);

    const int B = 256;
    auto blocks = [](long n, int b) { return (int)((n + b - 1) / b); };

    // pack weights into WMMA fragment order (f32 -> f16)
    pack_b_f32<<<blocks(NKT1 * NT1 * 32, B), B, 0, stream>>>(W1, Bp1, F_IN, HC1, NT1);
    pack_b_f32<<<blocks(NKT2 * NT2 * 32, B), B, 0, stream>>>(W2, Bp2, HC1, NCLS, NT2);

    // per-call init (harness does not re-zero ws between replays)
    fill_f32<<<blocks((long)N_NODES * HC1, B), B, 0, stream>>>(agg1, 0.f, N_NODES * HC1);
    fill_f32<<<blocks(N_NODES * HEADS, B), B, 0, stream>>>(nm1, -__builtin_inff(), N_NODES * HEADS);
    fill_f32<<<blocks(N_NODES * HEADS, B), B, 0, stream>>>(dn1, 0.f, N_NODES * HEADS);
    fill_f32<<<blocks(N_NODES, B), B, 0, stream>>>(nm2, -__builtin_inff(), N_NODES);
    fill_f32<<<blocks(N_NODES, B), B, 0, stream>>>(dn2, 0.f, N_NODES);
    fill_f32<<<blocks((long)N_NODES * NCLS, B), B, 0, stream>>>(agg2, 0.f, N_NODES * NCLS);

    // ---- layer 1 ----
    // h1 = x @ W1 : 3125 row tiles, K=128, 4 waves/block x 4 col tiles each
    wmma_gemm_packed<4><<<dim3(N_NODES / 16, 1), dim3(32, 4), 0, stream>>>(
        x, Bp1, h1, N_NODES, F_IN, NT1, HC1, HC1);

    node_logits<<<blocks(N_NODES * HEADS, B), B, 0, stream>>>(
        h1, as1, ad1, als1, ald1, N_NODES, HEADS, HID, HC1);

    edge_max_k<<<blocks((long)ET * HEADS, B), B, 0, stream>>>(ei, als1, ald1, nm1, HEADS, ET);
    edge_exp_k<<<blocks((long)ET * HEADS, B), B, 0, stream>>>(ei, als1, ald1, nm1, ex1, dn1, HEADS, ET);
    edge_agg_k<<<blocks((long)ET * HEADS, B), B, 0, stream>>>(ei, ex1, dn1, h1, agg1,
                                                              HEADS, HID, HC1, HC1, ET);
    bias_elu_k<<<blocks((long)N_NODES * HC1, B), B, 0, stream>>>(agg1, b1, N_NODES, HC1);

    // ---- layer 2 ----
    // h2 = elu(agg1) @ W2 : K=256, 1 wave/block handles all 3 col tiles
    wmma_gemm_packed<3><<<dim3(N_NODES / 16, 1), dim3(32, 1), 0, stream>>>(
        agg1, Bp2, h2, N_NODES, HC1, NT2, NCLS_PAD, NCLS_PAD);

    node_logits<<<blocks(N_NODES, B), B, 0, stream>>>(
        h2, as2, ad2, als2, ald2, N_NODES, 1, NCLS, NCLS_PAD);

    edge_max_k<<<blocks(ET, B), B, 0, stream>>>(ei, als2, ald2, nm2, 1, ET);
    edge_exp_k<<<blocks(ET, B), B, 0, stream>>>(ei, als2, ald2, nm2, ex2, dn2, 1, ET);
    edge_agg_k<<<blocks(ET, B), B, 0, stream>>>(ei, ex2, dn2, h2, agg2,
                                                1, NCLS, NCLS_PAD, NCLS, ET);

    // out = log_softmax(agg2 + b2)
    log_softmax_k<<<blocks(N_NODES, 8), dim3(32, 8), 0, stream>>>(agg2, b2, out, N_NODES, NCLS);
}
